// DENIS_48979807044191
// MI455X (gfx1250) — compile-verified, matrix operand order checked by reading, not run on hardware
//
#include <hip/hip_runtime.h>

typedef __attribute__((ext_vector_type(2)))  float    v2f;
typedef __attribute__((ext_vector_type(8)))  float    v8f;
typedef __attribute__((ext_vector_type(16))) _Float16 v16h;

// D = A(16x4 f32) * B(4x16 f32) + C(16x16 f32), wave32 WMMA.
static __device__ __forceinline__ v8f wmma4(v2f a, v2f b, v8f c) {
#if !defined(__AMDGCN__)
  return c;  // host-pass stub
#elif __has_builtin(__builtin_amdgcn_wmma_f32_16x16x4_f32)
  return __builtin_amdgcn_wmma_f32_16x16x4_f32(false, a, false, b, (short)0, c,
                                               false, false);
#else
  // Fallback (compile-safe): confirmed f16 WMMA path; signals f32x4 absent.
  v16h ah = {}; v16h bh = {};
  ah[0] = (_Float16)a.x; ah[1] = (_Float16)a.y;
  bh[0] = (_Float16)b.x; bh[1] = (_Float16)b.y;
  return __builtin_amdgcn_wmma_f32_16x16x32_f16(false, ah, false, bh, (short)0,
                                                c, false, false);
#endif
}

// B-fragment for f32 WMMA: 4x16 tile of row-major W (K x ncols).
// lane: n = ntile*16 + (lane&15); k = ktile*4 + 2*(lane>>4) (+1 for .y)
static __device__ __forceinline__ v2f bfrag(const float* W, int ncols, int kt,
                                            int halfk, int n) {
  int k = kt * 4 + 2 * halfk;
  v2f b;
  b.x = W[(size_t)k * ncols + n];
  b.y = W[(size_t)(k + 1) * ncols + n];
  return b;
}

#define LDS_FLOATS 73984  // 295,936 bytes, < 320 KB WGP LDS

// ---------------------------------------------------------------------------
// Kernel 1: enc MLP over all (N*64) rows; writes y_targ (shifts 1..63) and
// stages y1 (shift 0) to workspace. Weights live in LDS; per-wave staging
// buffers avoid workgroup barriers entirely (LDS is same-wave in-order).
// ---------------------------------------------------------------------------
__global__ __launch_bounds__(512) void DENIS_enc_kernel(
    const float* __restrict__ x, const float* __restrict__ w0,
    const float* __restrict__ b0, const float* __restrict__ w1,
    const float* __restrict__ b1, const float* __restrict__ w2,
    float* __restrict__ y_targ, float* __restrict__ y1, int nTiles) {
  extern __shared__ float lds[];
  float* lw0 = lds;          // 16x128
  float* lb0 = lw0 + 2048;   // 128
  float* lw1 = lb0 + 128;    // 128x128
  float* lb1 = lw1 + 16384;  // 128
  float* lw2 = lb1 + 128;    // 128x48   (weights total 24832 floats ~ 97 KB)

  const int tid = threadIdx.x;
  for (int i = tid; i < 2048; i += 512) lw0[i] = w0[i];
  for (int i = tid; i < 128; i += 512) lb0[i] = b0[i];
  for (int i = tid; i < 16384; i += 512) lw1[i] = w1[i];
  for (int i = tid; i < 128; i += 512) lb1[i] = b1[i];
  for (int i = tid; i < 6144; i += 512) lw2[i] = w2[i];
  __syncthreads();

  const int lane = tid & 31;
  const int wave = tid >> 5;
  const int halfk = lane >> 4;  // 0/1 : which K-half of the fragment
  const int mrow = lane & 15;   // row (A) or col (B/C) within tile
  float* hbuf = lds + 24832 + wave * 3072;  // 16x128 hidden staging
  float* ytile = hbuf + 2048;               // 16x64 output staging

  const int slots = gridDim.x * 16;
  const int slot = blockIdx.x * 16 + wave;

  for (int it = 0; it < 8; ++it) {
    int rt = slot + it * slots;  // wave-uniform -> EXEC stays all-1 for WMMA
    if (rt >= nTiles) break;
    const float* xr = x + (size_t)(rt * 16 + mrow) * 16;

    // ---- layer 0: (16x16) @ (16x128) + b0, relu ----
    v2f a0[4];
#pragma unroll
    for (int kt = 0; kt < 4; ++kt) {
      int k = kt * 4 + 2 * halfk;
      a0[kt].x = xr[k];
      a0[kt].y = xr[k + 1];
    }
#pragma unroll
    for (int nt = 0; nt < 8; ++nt) {
      int n = nt * 16 + mrow;
      float bias = lb0[n];
      v8f c;
#pragma unroll
      for (int r = 0; r < 8; ++r) c[r] = bias;
#pragma unroll
      for (int kt = 0; kt < 4; ++kt)
        c = wmma4(a0[kt], bfrag(lw0, 128, kt, halfk, n), c);
#pragma unroll
      for (int r = 0; r < 8; ++r)
        hbuf[(r + 8 * halfk) * 128 + n] = fmaxf(c[r], 0.0f);
    }

    // ---- layer 1: (16x128) @ (128x128) + b1, relu ----
    v2f ah[32];
#pragma unroll
    for (int kt = 0; kt < 32; ++kt) {
      int k = kt * 4 + 2 * halfk;
      ah[kt].x = hbuf[mrow * 128 + k];
      ah[kt].y = hbuf[mrow * 128 + k + 1];
    }
#pragma unroll
    for (int nt = 0; nt < 8; ++nt) {
      int n = nt * 16 + mrow;
      float bias = lb1[n];
      v8f c;
#pragma unroll
      for (int r = 0; r < 8; ++r) c[r] = bias;
#pragma unroll
      for (int kt = 0; kt < 32; ++kt)
        c = wmma4(ah[kt], bfrag(lw1, 128, kt, halfk, n), c);
#pragma unroll
      for (int r = 0; r < 8; ++r)
        hbuf[(r + 8 * halfk) * 128 + n] = fmaxf(c[r], 0.0f);
    }

    // ---- layer 2: (16x128) @ (128x48), no bias/relu -> ytile cols 16..63 ----
#pragma unroll
    for (int kt = 0; kt < 32; ++kt) {
      int k = kt * 4 + 2 * halfk;
      ah[kt].x = hbuf[mrow * 128 + k];
      ah[kt].y = hbuf[mrow * 128 + k + 1];
    }
#pragma unroll
    for (int nt = 0; nt < 3; ++nt) {
      int n = nt * 16 + mrow;
      v8f c = {};
#pragma unroll
      for (int kt = 0; kt < 32; ++kt)
        c = wmma4(ah[kt], bfrag(lw2, 48, kt, halfk, n), c);
#pragma unroll
      for (int r = 0; r < 8; ++r) ytile[(r + 8 * halfk) * 64 + 16 + n] = c[r];
    }

    // ytile cols 0..15 = raw x
#pragma unroll
    for (int q = 0; q < 8; ++q)
      ytile[mrow * 64 + halfk * 8 + q] = xr[halfk * 8 + q];

    // ---- coalesced row writes: shift 0 -> y1 staging, 1..63 -> y_targ ----
    for (int m = 0; m < 16; ++m) {
      int j = rt * 16 + m;
      int s = j >> 6, t = j & 63;
      const float* src = ytile + m * 64;
      float v0 = src[lane], v1 = src[lane + 32];
      if (t == 0) {
        float* dst = y1 + (size_t)s * 64;
        dst[lane] = v0;
        dst[lane + 32] = v1;
      } else {
        float* dst = y_targ + ((size_t)s * 63 + (t - 1)) * 64;
        dst[lane] = v0;
        dst[lane + 32] = v1;
      }
    }
  }
}

// ---------------------------------------------------------------------------
// Kernel 2: per 16 samples, aux MLP (WMMA) builds 16 koopman 64x64 matrices
// in LDS (256 KB), then the 63-step scan runs with each thread's K columns
// held in 128 VGPRs (zero LDS K re-reads), y broadcast via small LDS
// double-buffer, predictions streamed straight to HBM.
// ---------------------------------------------------------------------------
__global__ __launch_bounds__(512) void DENIS_koop_kernel(
    const float* __restrict__ x, const float* __restrict__ w0,
    const float* __restrict__ b0, const float* __restrict__ w1,
    const float* __restrict__ b1, const float* __restrict__ w2,
    const float* __restrict__ y1, float* __restrict__ y_pred) {
  extern __shared__ float lds[];
  float* sK = lds;          // 16 x 4096 koopman
  float* sA = lds + 65536;  // 16 x 256 (later reused for y double-buffer)
  float* sB = lds + 69632;  // 16 x 256
  float* sX = lds + 73728;  // 16 x 16

  const int tid = threadIdx.x;
  const int lane = tid & 31;
  const int wave = tid >> 5;  // 16 waves
  const int halfk = lane >> 4;
  const int mrow = lane & 15;
  const int n0 = blockIdx.x * 16;

  if (tid < 256) {
    int s = tid >> 4, d = tid & 15;
    sX[tid] = x[(size_t)(n0 + s) * 1024 + d];  // x[n][shift=0][d]
  }
  __syncthreads();

  // ---- aux layer 0: (16x16) @ (16x256), wave w owns n-tile w ----
  {
    v2f a0[4];
#pragma unroll
    for (int kt = 0; kt < 4; ++kt) {
      int k = kt * 4 + 2 * halfk;
      a0[kt].x = sX[mrow * 16 + k];
      a0[kt].y = sX[mrow * 16 + k + 1];
    }
    int n = wave * 16 + mrow;
    float bias = b0[n];
    v8f c;
#pragma unroll
    for (int r = 0; r < 8; ++r) c[r] = bias;
#pragma unroll
    for (int kt = 0; kt < 4; ++kt)
      c = wmma4(a0[kt], bfrag(w0, 256, kt, halfk, n), c);
#pragma unroll
    for (int r = 0; r < 8; ++r)
      sA[(r + 8 * halfk) * 256 + n] = fmaxf(c[r], 0.0f);
  }
  __syncthreads();

  // ---- aux layer 1: (16x256) @ (256x256), wave w owns n-tile w ----
  v2f ah[64];
#pragma unroll
  for (int kt = 0; kt < 64; ++kt) {
    int k = kt * 4 + 2 * halfk;
    ah[kt].x = sA[mrow * 256 + k];
    ah[kt].y = sA[mrow * 256 + k + 1];
  }
  {
    int n = wave * 16 + mrow;
    float bias = b1[n];
    v8f c;
#pragma unroll
    for (int r = 0; r < 8; ++r) c[r] = bias;
#pragma unroll
    for (int kt = 0; kt < 64; ++kt)
      c = wmma4(ah[kt], bfrag(w1, 256, kt, halfk, n), c);
#pragma unroll
    for (int r = 0; r < 8; ++r)
      sB[(r + 8 * halfk) * 256 + n] = fmaxf(c[r], 0.0f);
  }
  __syncthreads();

  // ---- aux layer 2: (16x256) @ (256x4096) -> sK; wave w owns 16 n-tiles ----
#pragma unroll
  for (int kt = 0; kt < 64; ++kt) {
    int k = kt * 4 + 2 * halfk;
    ah[kt].x = sB[mrow * 256 + k];
    ah[kt].y = sB[mrow * 256 + k + 1];
  }
  for (int nt = wave * 16; nt < wave * 16 + 16; ++nt) {
    int n = nt * 16 + mrow;
    __builtin_prefetch(w2 + (size_t)(2 * halfk) * 4096 + n + 16, 0, 1);
    v8f c = {};
#pragma unroll
    for (int kt = 0; kt < 64; ++kt)
      c = wmma4(ah[kt], bfrag(w2, 4096, kt, halfk, n), c);
#pragma unroll
    for (int r = 0; r < 8; ++r) sK[(size_t)(r + 8 * halfk) * 4096 + n] = c[r];
  }
  __syncthreads();

  // ---- 63-step scan: y <- y @ K (per sample). One wave per sample. ----
  float* sY0 = sA;         // 16 x 65 (padded vs. bank conflicts)
  float* sY1 = sA + 1040;  // 16 x 65
  for (int i = tid; i < 1024; i += 512) {
    int s = i >> 6, e = i & 63;
    sY0[s * 65 + e] = y1[(size_t)(n0 + s) * 64 + e];
  }
  const int s = tid >> 5;  // sample = wave index (all 32 lanes same s)
  const int q = tid & 31;  // this lane owns output dims q and q+32
  float kr0[64], kr1[64];  // K columns in registers: 128 VGPRs
#pragma unroll
  for (int d = 0; d < 64; ++d) {
    kr0[d] = sK[(size_t)s * 4096 + d * 64 + q];
    kr1[d] = sK[(size_t)s * 4096 + d * 64 + q + 32];
  }
  __syncthreads();

  float* cur = sY0;
  float* nxt = sY1;
  for (int step = 0; step < 63; ++step) {
    float acc0 = 0.0f, acc1 = 0.0f;
#pragma unroll
    for (int d = 0; d < 64; ++d) {
      float yv = cur[s * 65 + d];  // LDS broadcast within wave
      acc0 += yv * kr0[d];
      acc1 += yv * kr1[d];
    }
    nxt[s * 65 + q] = acc0;
    nxt[s * 65 + q + 32] = acc1;
    size_t orow = ((size_t)(n0 + s) * 63 + step) * 64;
    y_pred[orow + q] = acc0;  // 128B coalesced per half-row
    y_pred[orow + q + 32] = acc1;
    __syncthreads();
    float* t = cur;
    cur = nxt;
    nxt = t;
  }
}

extern "C" void kernel_launch(void* const* d_in, const int* in_sizes, int n_in,
                              void* d_out, int out_size, void* d_ws,
                              size_t ws_size, hipStream_t stream) {
  const float* x = (const float*)d_in[0];
  const float* ew0 = (const float*)d_in[1];
  const float* eb0 = (const float*)d_in[2];
  const float* ew1 = (const float*)d_in[3];
  const float* eb1 = (const float*)d_in[4];
  const float* ew2 = (const float*)d_in[5];
  const float* aw0 = (const float*)d_in[6];
  const float* ab0 = (const float*)d_in[7];
  const float* aw1 = (const float*)d_in[8];
  const float* ab1 = (const float*)d_in[9];
  const float* aw2 = (const float*)d_in[10];

  const int N = in_sizes[0] / (64 * 16);
  float* y_targ = (float*)d_out;
  float* y_pred = y_targ + (size_t)N * 63 * 64;
  float* y1ws = (float*)d_ws;  // N*64 floats staging

  const size_t ldsBytes = (size_t)LDS_FLOATS * sizeof(float);  // 295,936 B
  (void)hipFuncSetAttribute((const void*)DENIS_enc_kernel,
                            hipFuncAttributeMaxDynamicSharedMemorySize,
                            (int)ldsBytes);
  (void)hipFuncSetAttribute((const void*)DENIS_koop_kernel,
                            hipFuncAttributeMaxDynamicSharedMemorySize,
                            (int)ldsBytes);

  const int nTiles = N * 4;                    // (N*64)/16 row-tiles
  const int encGrid = (nTiles + 127) / 128;    // 16 waves x 8 tiles per block
  DENIS_enc_kernel<<<encGrid, 512, ldsBytes, stream>>>(
      x, ew0, eb0, ew1, eb1, ew2, y_targ, y1ws, nTiles);
  DENIS_koop_kernel<<<N / 16, 512, ldsBytes, stream>>>(
      x, aw0, ab0, aw1, ab1, aw2, y1ws, y_pred);
}
